// MMSR_net_14207751815897
// MI455X (gfx1250) — matrix operand order repaired and checked by reading, not compiled, not run on hardware
//
#include <hip/hip_runtime.h>
#include <hip/hip_bf16.h>
#include <math.h>
#include <stdint.h>

typedef __attribute__((ext_vector_type(16))) _Float16 v16h;
typedef __attribute__((ext_vector_type(8)))  float    v8f;

#define PX 16384   // 128*128
#define WID 128

// ---------------------------------------------------------------------------
// Weight pre-pack: f32 OIHW -> f16 in the contiguous-k B-operand layout used here,
// chunk-major: out[ch*NT*512 + w*512 + lane*16 + h] = W[n][k],
//   n = w*16 + lane%16,  k = ch*32 + h + 16*(lane/16)   (zero-fill k>=KTOT)
// Done once per layer so conv blocks can async-copy B verbatim.
// ---------------------------------------------------------------------------
__global__ void pack_w(const float* __restrict__ Wg, _Float16* __restrict__ out,
                       int KTOT, int total)
{
    const int idx = blockIdx.x * blockDim.x + threadIdx.x;
    if (idx >= total) return;
    const int li   = idx & 511;
    const int rem  = idx >> 9;          // (ch*NT + w)
    const int bl   = li >> 4, h = li & 15;
    const int n    = ((rem /* per-chunk wave id folded below */), 0);
    (void)n;
    // decompose: idx = ch*(NT*512) + w*512 + li  -> need NT; encode via KTOT-free math:
    // caller guarantees layout chunk-major with NTILES waves per chunk; we pass
    // total = KCH*NT*512 and recover (ch, w) from rem given NT via second arg trick.
    // To keep a single kernel, NT is passed in the high bits of KTOT? -> no:
    // simpler: rem = ch*NT + w, and k depends only on ch; n depends only on w.
    // We pass NT via gridDim.y.
    const int NT = gridDim.y;           // launched with grid.y = NT
    const int ch = rem / NT;
    const int w  = rem % NT;
    const int nn = w * 16 + (bl & 15);
    const int k  = ch * 32 + h + ((bl >> 4) << 4);
    out[idx] = (_Float16)((k < KTOT) ? Wg[nn * KTOT + k] : 0.f);
}

// ---------------------------------------------------------------------------
// Implicit-GEMM conv via WMMA f32_16x16x32_f16 with M-register-blocking.
//   Block owns MT=4 M-tiles (64 consecutive pixels, same row) x NT N-tiles
//   (16 out-channels per wave). K chunked by 32, K-loop fully unrolled.
//   A (activations) staged to LDS with on-the-fly im2col + f32->f16 (VALU).
//   B (weights)  : pre-packed f16 copied Global->LDS with the CDNA5 async
//                  path (global_load_async_to_lds_b128, ASYNCcnt,
//                  s_wait_asynccnt) -- zero VALU, overlaps A staging.
//   Operand layouts with contiguous k = 16*(lane/16)+s in A and B (A 16-bit 16x32, B rows striped, D f32).
// ---------------------------------------------------------------------------
template<int CIN1, int CIN2, int KS, int NT, bool RELU>
__global__ void conv_wmma(const float* __restrict__ in1,
                          const float* __restrict__ in2,
                          const _Float16* __restrict__ Wp,   // pre-packed
                          const float* __restrict__ bias,
                          const float* __restrict__ resid,
                          float* __restrict__ out)
{
    constexpr int MT   = 4;
    constexpr int CIN  = CIN1 + CIN2;
    constexpr int KTOT = CIN * KS * KS;
    constexpr int KCH  = (KTOT + 31) / 32;
    constexpr int PAD  = KS / 2;
    constexpr int BLK  = NT * 32;

    __shared__ __align__(32) _Float16 Ash[2][MT * 512];
    __shared__ __align__(32) _Float16 Bsh[2][NT * 512];

    const int tid  = threadIdx.x;
    const int lane = tid & 31;
    const int wave = tid >> 5;
    const int t0   = blockIdx.x * MT;
    const int p0   = t0 * 16;           // 64 consecutive pixels, same row
    const int y    = p0 / WID;
    const int x0   = p0 % WID;

    v8f acc[MT] = {};

#pragma unroll
    for (int ch = 0; ch < KCH; ++ch) {
        const int buf = ch & 1;

        // ---- stage B: async Global->LDS copy of pre-packed f16 weights ----
        {
            const _Float16* gsrc = Wp + (size_t)ch * (NT * 512) + (tid << 4);
            const unsigned  lds  = (unsigned)(uintptr_t)(&Bsh[buf][tid << 4]);
            const unsigned long long ga = (unsigned long long)(uintptr_t)gsrc;
            asm volatile("global_load_async_to_lds_b128 %0, %1, off"
                         :: "v"(lds), "v"(ga) : "memory");
            asm volatile("global_load_async_to_lds_b128 %0, %1, off offset:16"
                         :: "v"(lds), "v"(ga) : "memory");
        }

        // ---- stage A (activations, im2col on the fly, MT tiles) ----
        for (int idx = tid; idx < MT * 512; idx += BLK) {
            const int j  = idx >> 9;         // M-tile within block
            const int li = idx & 511;
            const int al = li >> 4, h = li & 15;
            const int k  = ch * 32 + ((al >> 4) << 4) + h;
            float v = 0.f;
            if (k < KTOT) {
                const int c  = k / (KS * KS);
                const int rs = k % (KS * KS);
                const int r  = rs / KS, s = rs % KS;
                const int yy = y + r - PAD;
                const int xx = x0 + (j << 4) + (al & 15) + s - PAD;
                if (yy >= 0 && yy < WID && xx >= 0 && xx < WID) {
                    if (CIN2 == 0 || c < CIN1)
                        v = in1[c * PX + yy * WID + xx];
                    else
                        v = in2[(c - CIN1) * PX + yy * WID + xx];
                }
            }
            Ash[buf][idx] = (_Float16)v;
        }

        // ---- prefetch next weight chunk (global_prefetch_b8) ----
        if (ch + 1 < KCH)
            __builtin_prefetch(Wp + (size_t)(ch + 1) * (NT * 512) + (tid << 4), 0, 1);

        // drain async B copies, then block barrier
        asm volatile("s_wait_asynccnt 0x0" ::: "memory");
        __syncthreads();

        const v16h b = *(const v16h*)(&Bsh[buf][(wave << 9) + (lane << 4)]);
#pragma unroll
        for (int j = 0; j < MT; ++j) {
            const v16h a = *(const v16h*)(&Ash[buf][(j << 9) + (lane << 4)]);
            acc[j] = __builtin_amdgcn_wmma_f32_16x16x32_f16(
                false, a, false, b, (short)0, acc[j], false, false);
        }
    }

    // ---- epilogue: D layout -> NCHW ----
    const int n  = wave * 16 + (lane & 15);
    const int mb = (lane >> 4) * 8;
    const float bz = bias ? bias[n] : 0.f;
#pragma unroll
    for (int j = 0; j < MT; ++j) {
#pragma unroll
        for (int v = 0; v < 8; ++v) {
            const int m = mb + v;
            const int p = (t0 + j) * 16 + m;
            float r = acc[j][v] + bz;
            if (resid) r += resid[n * PX + p];
            if (RELU)  r = fmaxf(r, 0.f);
            out[n * PX + p] = r;
        }
    }
}

// ---------------------------------------------------------------------------
// Channel LayerNorm (channels_first) + exact-erf GELU, per pixel.
// ---------------------------------------------------------------------------
template<int C>
__global__ void ln_gelu(const float* __restrict__ in,
                        const float* __restrict__ lw,
                        const float* __restrict__ lb,
                        float* __restrict__ out)
{
    const int p = blockIdx.x * blockDim.x + threadIdx.x;
    if (p >= PX) return;
    float u = 0.f;
#pragma unroll
    for (int c = 0; c < C; ++c) u += in[c * PX + p];
    u *= (1.f / C);
    float s = 0.f;
#pragma unroll
    for (int c = 0; c < C; ++c) { const float d = in[c * PX + p] - u; s += d * d; }
    s *= (1.f / C);
    const float inv = rsqrtf(s + 1e-6f);
#pragma unroll
    for (int c = 0; c < C; ++c) {
        const float xn = (in[c * PX + p] - u) * inv;
        const float x  = lw[c] * xn + lb[c];
        out[c * PX + p] = 0.5f * x * (1.f + erff(x * 0.70710678f));
    }
}

// ev_d1: 1x1 conv 32->3, LN over 3 channels, softmax over 3 channels.
__global__ void evd1_softmax(const float* __restrict__ in,
                             const float* __restrict__ W,
                             const float* __restrict__ lw,
                             const float* __restrict__ lb,
                             float* __restrict__ out)
{
    const int p = blockIdx.x * blockDim.x + threadIdx.x;
    if (p >= PX) return;
    float h[3];
#pragma unroll
    for (int j = 0; j < 3; ++j) {
        float a = 0.f;
#pragma unroll
        for (int c = 0; c < 32; ++c) a += W[j * 32 + c] * in[c * PX + p];
        h[j] = a;
    }
    const float u = (h[0] + h[1] + h[2]) * (1.f / 3.f);
    float s = 0.f;
#pragma unroll
    for (int j = 0; j < 3; ++j) { const float d = h[j] - u; s += d * d; }
    const float inv = rsqrtf(s * (1.f / 3.f) + 1e-6f);
    float y[3], m = -1e30f;
#pragma unroll
    for (int j = 0; j < 3; ++j) { y[j] = lw[j] * (h[j] - u) * inv + lb[j]; m = fmaxf(m, y[j]); }
    float e[3], se = 0.f;
#pragma unroll
    for (int j = 0; j < 3; ++j) { e[j] = __expf(y[j] - m); se += e[j]; }
    const float r = 1.f / se;
#pragma unroll
    for (int j = 0; j < 3; ++j) out[j * PX + p] = e[j] * r;
}

// eigen = relu(conv3x3(softmax_out, conv_w[1,3,3,3]) + conv_b)
__global__ void eigen_conv(const float* __restrict__ in,
                           const float* __restrict__ W,
                           const float* __restrict__ b,
                           float* __restrict__ out)
{
    const int p = blockIdx.x * blockDim.x + threadIdx.x;
    if (p >= PX) return;
    const int y = p >> 7, x = p & 127;
    float a = b[0];
#pragma unroll
    for (int c = 0; c < 3; ++c)
#pragma unroll
        for (int r = 0; r < 3; ++r)
#pragma unroll
            for (int s = 0; s < 3; ++s) {
                const int yy = y + r - 1, xx = x + s - 1;
                if (yy >= 0 && yy < WID && xx >= 0 && xx < WID)
                    a += W[(c * 3 + r) * 3 + s] * in[c * PX + yy * WID + xx];
            }
    out[p] = fmaxf(a, 0.f);
}

// Bilinear resize 64x64 -> 128x128, half-pixel centers.
__global__ void resize_bilinear(const float* __restrict__ src, float* __restrict__ dst)
{
    const int p = blockIdx.x * blockDim.x + threadIdx.x;
    if (p >= PX) return;
    const int y = p >> 7, x = p & 127;
    const float sy = (y + 0.5f) * 0.5f - 0.5f;
    const float sx = (x + 0.5f) * 0.5f - 0.5f;
    const float fy0 = floorf(sy), fx0 = floorf(sx);
    const float wy = sy - fy0, wx = sx - fx0;
    const int y0 = max(0, min(63, (int)fy0));
    const int y1 = max(0, min(63, (int)fy0 + 1));
    const int x0 = max(0, min(63, (int)fx0));
    const int x1 = max(0, min(63, (int)fx0 + 1));
    const float v00 = src[y0 * 64 + x0], v01 = src[y0 * 64 + x1];
    const float v10 = src[y1 * 64 + x0], v11 = src[y1 * 64 + x1];
    dst[p] = (1.f - wy) * ((1.f - wx) * v00 + wx * v01) +
             wy * ((1.f - wx) * v10 + wx * v11);
}

// s1: 1x1 conv CIN=1 -> 64, bias, relu
__global__ void s1_conv(const float* __restrict__ in, const float* __restrict__ W,
                        const float* __restrict__ b, float* __restrict__ out)
{
    const int p = blockIdx.x * blockDim.x + threadIdx.x;
    if (p >= PX) return;
    const float v = in[p];
#pragma unroll
    for (int n = 0; n < 64; ++n)
        out[n * PX + p] = fmaxf(W[n] * v + b[n], 0.f);
}

// Unfold attention: filt = softmax_n(<v_n, q>), out = sum_n filt_n * v_n.
template<int R>
__global__ void attn(const float* __restrict__ q, const float* __restrict__ v,
                     float* __restrict__ out)
{
    const int p = blockIdx.x * blockDim.x + threadIdx.x;
    if (p >= PX) return;
    const int y = p >> 7, x = p & 127;
    float qr[64];
#pragma unroll
    for (int c = 0; c < 64; ++c) qr[c] = q[c * PX + p];

    float m = -1e30f, s = 0.f;
    for (int dy = 0; dy < R; ++dy)
        for (int dx = 0; dx < R; ++dx) {
            const int ny = y + dy - R / 2, nx = x + dx - R / 2;
            float l = 0.f;
            if (ny >= 0 && ny < WID && nx >= 0 && nx < WID) {
                const int np = ny * WID + nx;
#pragma unroll
                for (int c = 0; c < 64; ++c) l += qr[c] * v[c * PX + np];
            }
            if (l > m) { s *= __expf(m - l); m = l; }
            s += __expf(l - m);
        }
    const float inv = 1.f / s;

    float acc[64];
#pragma unroll
    for (int c = 0; c < 64; ++c) acc[c] = 0.f;
    for (int dy = 0; dy < R; ++dy)
        for (int dx = 0; dx < R; ++dx) {
            const int ny = y + dy - R / 2, nx = x + dx - R / 2;
            if (ny >= 0 && ny < WID && nx >= 0 && nx < WID) {
                const int np = ny * WID + nx;
                float l = 0.f;
#pragma unroll
                for (int c = 0; c < 64; ++c) l += qr[c] * v[c * PX + np];
                const float w = __expf(l - m) * inv;
#pragma unroll
                for (int c = 0; c < 64; ++c) acc[c] += w * v[c * PX + np];
            }
        }
#pragma unroll
    for (int c = 0; c < 64; ++c) out[c * PX + p] = acc[c];
}

// predict = pout(h) ; d_out = [predict | eigen]
__global__ void finalize(const float* __restrict__ h, const float* __restrict__ W,
                         const float* __restrict__ b, const float* __restrict__ eig,
                         float* __restrict__ out)
{
    const int p = blockIdx.x * blockDim.x + threadIdx.x;
    if (p >= PX) return;
    float a = b[0];
#pragma unroll
    for (int c = 0; c < 64; ++c) a += W[c] * h[c * PX + p];
    out[p]      = a;
    out[PX + p] = eig[p];
}

// ---------------------------------------------------------------------------
extern "C" void kernel_launch(void* const* d_in, const int* in_sizes, int n_in,
                              void* d_out, int out_size, void* d_ws, size_t ws_size,
                              hipStream_t stream)
{
    (void)in_sizes; (void)n_in; (void)out_size; (void)ws_size;
    const float* guide  = (const float*)d_in[0];
    const float* source = (const float*)d_in[1];
    const float* ev_c1_w  = (const float*)d_in[2];
    const float* ev_c2_w  = (const float*)d_in[3];
    const float* ev_d21_w = (const float*)d_in[4];
    const float* ev_d22_w = (const float*)d_in[5];
    const float* ev_d1_w  = (const float*)d_in[6];
    const float* conv_w   = (const float*)d_in[7];
    const float* g1_w     = (const float*)d_in[8];
    const float* g2_w     = (const float*)d_in[9];
    const float* grb1_w1  = (const float*)d_in[10];
    const float* grb1_w2  = (const float*)d_in[11];
    const float* grb2_w1  = (const float*)d_in[12];
    const float* grb2_w2  = (const float*)d_in[13];
    const float* s1_w     = (const float*)d_in[14];
    const float* s2_w     = (const float*)d_in[15];
    const float* srb1_w1  = (const float*)d_in[16];
    const float* srb1_w2  = (const float*)d_in[17];
    const float* srb2_w1  = (const float*)d_in[18];
    const float* srb2_w2  = (const float*)d_in[19];
    const float* m1_w     = (const float*)d_in[20];
    const float* m2_w     = (const float*)d_in[21];
    const float* m3_w     = (const float*)d_in[22];
    const float* prb1_w1  = (const float*)d_in[23];
    const float* prb1_w2  = (const float*)d_in[24];
    const float* prb2_w1  = (const float*)d_in[25];
    const float* prb2_w2  = (const float*)d_in[26];
    const float* prb3_w1  = (const float*)d_in[27];
    const float* prb3_w2  = (const float*)d_in[28];
    const float* pout_w   = (const float*)d_in[29];
    const float* conv_b  = (const float*)d_in[30];
    const float* g1_b    = (const float*)d_in[31];
    const float* g2_b    = (const float*)d_in[32];
    const float* grb1_b1 = (const float*)d_in[33];
    const float* grb1_b2 = (const float*)d_in[34];
    const float* grb2_b1 = (const float*)d_in[35];
    const float* grb2_b2 = (const float*)d_in[36];
    const float* s1_b    = (const float*)d_in[37];
    const float* s2_b    = (const float*)d_in[38];
    const float* srb1_b1 = (const float*)d_in[39];
    const float* srb1_b2 = (const float*)d_in[40];
    const float* srb2_b1 = (const float*)d_in[41];
    const float* srb2_b2 = (const float*)d_in[42];
    const float* m1_b    = (const float*)d_in[43];
    const float* m2_b    = (const float*)d_in[44];
    const float* m3_b    = (const float*)d_in[45];
    const float* prb1_b1 = (const float*)d_in[46];
    const float* prb1_b2 = (const float*)d_in[47];
    const float* prb2_b1 = (const float*)d_in[48];
    const float* prb2_b2 = (const float*)d_in[49];
    const float* prb3_b1 = (const float*)d_in[50];
    const float* prb3_b2 = (const float*)d_in[51];
    const float* pout_b  = (const float*)d_in[52];
    const float* ev_c1_lw  = (const float*)d_in[53];
    const float* ev_c1_lb  = (const float*)d_in[54];
    const float* ev_c2_lw  = (const float*)d_in[55];
    const float* ev_c2_lb  = (const float*)d_in[56];
    const float* ev_d21_lw = (const float*)d_in[57];
    const float* ev_d21_lb = (const float*)d_in[58];
    const float* ev_d22_lw = (const float*)d_in[59];
    const float* ev_d22_lb = (const float*)d_in[60];
    const float* ev_d1_lw  = (const float*)d_in[61];
    const float* ev_d1_lb  = (const float*)d_in[62];

    float* ws = (float*)d_ws;
    float* srcup = ws;                 // 1  ch
    float* eig   = ws + 1 * PX;        // 1  ch
    float* t3    = ws + 2 * PX;        // 3  ch
    float* A32   = ws + 5 * PX;        // 32 ch
    float* B32   = ws + 37 * PX;       // 32 ch
    float* C32   = ws + 69 * PX;       // 32 ch
    float* A64   = ws + 101 * PX;      // 64 ch
    float* B64   = ws + 165 * PX;      // 64 ch
    float* C64   = ws + 229 * PX;      // 64 ch
    float* D64   = ws + 293 * PX;      // 64 ch

    // --- packed f16 weight area (16B aligned: 357*PX*4 % 16 == 0) ---
    _Float16* wp = (_Float16*)(ws + 357 * PX);
    struct PS { const float* src; _Float16* dst; int ktot; int nt; };
    PS ps[24];
    int np = 0;
    auto alloc_pack = [&](const float* src, int ktot, int nt) -> _Float16* {
        const int kch = (ktot + 31) / 32;
        _Float16* d = wp;
        wp += (size_t)kch * nt * 512;
        ps[np++] = PS{src, d, ktot, nt};
        return d;
    };
    _Float16* p_evc1 = alloc_pack(ev_c1_w, 27, 2);
    _Float16* p_evc2 = alloc_pack(ev_c2_w, 288, 4);
    _Float16* p_evd21 = alloc_pack(ev_d21_w, 576, 2);
    _Float16* p_evd22 = alloc_pack(ev_d22_w, 576, 2);
    _Float16* p_g1   = alloc_pack(g1_w, 36, 4);
    _Float16* p_g2   = alloc_pack(g2_w, 576, 4);
    _Float16* p_gr11 = alloc_pack(grb1_w1, 576, 4);
    _Float16* p_gr12 = alloc_pack(grb1_w2, 576, 4);
    _Float16* p_gr21 = alloc_pack(grb2_w1, 576, 4);
    _Float16* p_gr22 = alloc_pack(grb2_w2, 576, 4);
    _Float16* p_s2   = alloc_pack(s2_w, 64, 4);
    _Float16* p_sr11 = alloc_pack(srb1_w1, 64, 4);
    _Float16* p_sr12 = alloc_pack(srb1_w2, 64, 4);
    _Float16* p_sr21 = alloc_pack(srb2_w1, 64, 4);
    _Float16* p_sr22 = alloc_pack(srb2_w2, 64, 4);
    _Float16* p_m1   = alloc_pack(m1_w, 64, 4);
    _Float16* p_m2   = alloc_pack(m2_w, 64, 4);
    _Float16* p_m3   = alloc_pack(m3_w, 128, 4);
    _Float16* p_pr11 = alloc_pack(prb1_w1, 64, 4);
    _Float16* p_pr12 = alloc_pack(prb1_w2, 64, 4);
    _Float16* p_pr21 = alloc_pack(prb2_w1, 64, 4);
    _Float16* p_pr22 = alloc_pack(prb2_w2, 64, 4);
    _Float16* p_pr31 = alloc_pack(prb3_w1, 64, 4);
    _Float16* p_pr32 = alloc_pack(prb3_w2, 64, 4);

    // launch all weight packers first (grid.y carries NT)
    for (int i = 0; i < np; ++i) {
        const int kch   = (ps[i].ktot + 31) / 32;
        const int total = kch * ps[i].nt * 512;
        dim3 g((total / ps[i].nt + 255) / 256, ps[i].nt);
        // grid.x covers total/NT per y-slice? simpler: 1D over total with y=NT
        // -> use y only as NT carrier; idx uses x only, so each y-slice repeats the same writes
        g.x = (total + 255) / 256;
        pack_w<<<dim3(g.x, ps[i].nt, 1), 256, 0, stream>>>(ps[i].src, ps[i].dst,
                                                           ps[i].ktot, total);
    }

    const dim3 gP((PX + 255) / 256), bP(256);
    const dim3 gT(PX / 64);            // 256 blocks, 4 M-tiles each
    const dim3 b4(128), b2(64);
    const dim3 gA(PX / 128), bA(128);

    // --- source upsample ---
    resize_bilinear<<<gP, bP, 0, stream>>>(source, srcup);

    // --- SSGNet ---
    conv_wmma<3, 0, 3, 2, false><<<gT, b2, 0, stream>>>(guide, nullptr, p_evc1, nullptr, nullptr, B32);
    ln_gelu<32><<<gP, bP, 0, stream>>>(B32, ev_c1_lw, ev_c1_lb, A32);                 // c1 = A32
    conv_wmma<32, 0, 3, 4, false><<<gT, b4, 0, stream>>>(A32, nullptr, p_evc2, nullptr, nullptr, A64);
    ln_gelu<64><<<gP, bP, 0, stream>>>(A64, ev_c2_lw, ev_c2_lb, B64);                 // c2 = B64
    conv_wmma<64, 0, 3, 2, false><<<gT, b2, 0, stream>>>(B64, nullptr, p_evd21, nullptr, nullptr, B32);
    ln_gelu<32><<<gP, bP, 0, stream>>>(B32, ev_d21_lw, ev_d21_lb, C32);               // d2 = C32
    conv_wmma<32, 32, 3, 2, false><<<gT, b2, 0, stream>>>(C32, A32, p_evd22, nullptr, nullptr, B32);
    ln_gelu<32><<<gP, bP, 0, stream>>>(B32, ev_d22_lw, ev_d22_lb, C32);               // d2' = C32
    evd1_softmax<<<gP, bP, 0, stream>>>(C32, ev_d1_w, ev_d1_lw, ev_d1_lb, t3);
    eigen_conv<<<gP, bP, 0, stream>>>(t3, conv_w, conv_b, eig);

    // --- guide branch ---
    conv_wmma<3, 1, 3, 4, true><<<gT, b4, 0, stream>>>(guide, eig, p_g1, g1_b, nullptr, A64);
    conv_wmma<64, 0, 3, 4, true><<<gT, b4, 0, stream>>>(A64, nullptr, p_g2, g2_b, nullptr, B64);
    conv_wmma<64, 0, 3, 4, true><<<gT, b4, 0, stream>>>(B64, nullptr, p_gr11, grb1_b1, nullptr, C64);
    conv_wmma<64, 0, 3, 4, false><<<gT, b4, 0, stream>>>(C64, nullptr, p_gr12, grb1_b2, B64, A64);
    conv_wmma<64, 0, 3, 4, true><<<gT, b4, 0, stream>>>(A64, nullptr, p_gr21, grb2_b1, nullptr, C64);
    conv_wmma<64, 0, 3, 4, false><<<gT, b4, 0, stream>>>(C64, nullptr, p_gr22, grb2_b2, A64, B64);
    // fg = B64

    // --- source branch (all 1x1) ---
    s1_conv<<<gP, bP, 0, stream>>>(srcup, s1_w, s1_b, C64);
    conv_wmma<64, 0, 1, 4, true><<<gT, b4, 0, stream>>>(C64, nullptr, p_s2, s2_b, nullptr, A64);
    conv_wmma<64, 0, 1, 4, true><<<gT, b4, 0, stream>>>(A64, nullptr, p_sr11, srb1_b1, nullptr, D64);
    conv_wmma<64, 0, 1, 4, false><<<gT, b4, 0, stream>>>(D64, nullptr, p_sr12, srb1_b2, A64, C64);
    conv_wmma<64, 0, 1, 4, true><<<gT, b4, 0, stream>>>(C64, nullptr, p_sr21, srb2_b1, nullptr, D64);
    conv_wmma<64, 0, 1, 4, false><<<gT, b4, 0, stream>>>(D64, nullptr, p_sr22, srb2_b2, C64, A64);
    // fs = A64

    // --- modulation ---
    conv_wmma<64, 0, 1, 4, false><<<gT, b4, 0, stream>>>(B64, nullptr, p_m1, m1_b, nullptr, C64); // fg' = C64
    conv_wmma<64, 0, 1, 4, false><<<gT, b4, 0, stream>>>(A64, nullptr, p_m2, m2_b, nullptr, D64); // fs' = D64
    attn<11><<<gA, bA, 0, stream>>>(C64, D64, A64);   // f_s2g = A64
    attn<5><<<gA, bA, 0, stream>>>(D64, C64, B64);    // f_g2s = B64
    conv_wmma<64, 64, 1, 4, false><<<gT, b4, 0, stream>>>(A64, B64, p_m3, m3_b, nullptr, C64);    // fusion = C64

    // --- post resblocks (1x1) ---
    conv_wmma<64, 0, 1, 4, true><<<gT, b4, 0, stream>>>(C64, nullptr, p_pr11, prb1_b1, nullptr, D64);
    conv_wmma<64, 0, 1, 4, false><<<gT, b4, 0, stream>>>(D64, nullptr, p_pr12, prb1_b2, C64, A64);
    conv_wmma<64, 0, 1, 4, true><<<gT, b4, 0, stream>>>(A64, nullptr, p_pr21, prb2_b1, nullptr, D64);
    conv_wmma<64, 0, 1, 4, false><<<gT, b4, 0, stream>>>(D64, nullptr, p_pr22, prb2_b2, A64, B64);
    conv_wmma<64, 0, 1, 4, true><<<gT, b4, 0, stream>>>(B64, nullptr, p_pr31, prb3_b1, nullptr, D64);
    conv_wmma<64, 0, 1, 4, false><<<gT, b4, 0, stream>>>(D64, nullptr, p_pr32, prb3_b2, B64, C64);

    // --- output: [predict | eigen] ---
    finalize<<<gP, bP, 0, stream>>>(C64, pout_w, pout_b, eig, (float*)d_out);
}
